// MultiHeadAttentionV2_73512660238555
// MI455X (gfx1250) — matrix operand order repaired and checked by reading, not compiled, not run on hardware
//
#include <hip/hip_runtime.h>
#include <hip/hip_bf16.h>

// ---------------------------------------------------------------------------
// MultiHeadAttentionV2 forward for MI455X (gfx1250, wave32, WMMA).
//   * QKV projections: WMMA GEMMs reading f32 activations/weights directly
//     (f32->f16 convert inside the fragment loaders; no cast round-trips).
//   * Fused attention: QK^T (WMMA) -> softmax (256KB LDS score tile, values
//     register-resident) -> PV (WMMA with P fragments from LDS, dual
//     accumulators for ILP). The 512MB attn tensor is written to HBM once.
//   * FC GEMM (+bias+residual) -> LayerNorm.
//   * Branch-free loop bodies; startup global_prefetch_b8 only.
// ---------------------------------------------------------------------------

typedef _Float16 half_t;
typedef __attribute__((ext_vector_type(16))) _Float16 v16h;
typedef __attribute__((ext_vector_type(8)))  _Float16 v8h;
typedef __attribute__((ext_vector_type(8)))  float    v8f;
typedef __attribute__((ext_vector_type(4)))  float    v4f;

#define B_SZ   2
#define L_SZ   2048
#define T_SZ   2048
#define H_SZ   16
#define DK_SZ  64
#define DM_SZ  1024
#define MROWS  (B_SZ * L_SZ)           // 4096
#define TILE_L 32                      // l-rows per fused-attention block
#define SROW   2052                    // padded LDS row stride (floats)

// ---------------- WMMA helpers -------------------------------------------

__device__ __forceinline__ v8f wmma_f16(v16h a, v16h b, v8f c) {
  // D = A(16x32) x B(32x16) + C, f32 accumulate
  return __builtin_amdgcn_wmma_f32_16x16x32_f16(
      /*neg_a=*/false, a, /*neg_b=*/false, b,
      /*c_mod=*/(short)0, c, /*reuse_a=*/false, /*reuse_b=*/false);
}

// A fragment (16x32, row-major src). lanes 0-15: rows m=lane, halves 0..7 =
// K[kb..kb+7], halves 8..15 = K[16+kb..16+kb+7], kb = 8*(lane>=16).
__device__ __forceinline__ v16h load_frag_a(const half_t* p0, int ld, int lane) {
  const int m  = lane & 15;
  const int kb = (lane >> 4) << 3;            // 0 or 8
  const half_t* p = p0 + (size_t)m * (size_t)ld + kb;
  v8h lo = *reinterpret_cast<const v8h*>(p);
  v8h hi = *reinterpret_cast<const v8h*>(p + 16);
  v16h f;
#pragma unroll
  for (int i = 0; i < 8; ++i) { f[i] = lo[i]; f[i + 8] = hi[i]; }
  return f;
}

// Same fragment, f32 source converted in-register (v_cvt under load shadow).
__device__ __forceinline__ v16h load_frag_a(const float* p0, int ld, int lane) {
  const int m  = lane & 15;
  const int kb = (lane >> 4) << 3;
  const float* p = p0 + (size_t)m * (size_t)ld + kb;
  v4f x0 = *reinterpret_cast<const v4f*>(p);
  v4f x1 = *reinterpret_cast<const v4f*>(p + 4);
  v4f x2 = *reinterpret_cast<const v4f*>(p + 16);
  v4f x3 = *reinterpret_cast<const v4f*>(p + 20);
  v16h f;
#pragma unroll
  for (int i = 0; i < 4; ++i) {
    f[i]      = (half_t)x0[i];
    f[4 + i]  = (half_t)x1[i];
    f[8 + i]  = (half_t)x2[i];
    f[12 + i] = (half_t)x3[i];
  }
  return f;
}

// A fragment sourced from the LDS score tile (f32 -> f16 convert).
__device__ __forceinline__ v16h load_frag_a_lds(const float* pr) {
  v4f x0 = *reinterpret_cast<const v4f*>(pr);
  v4f x1 = *reinterpret_cast<const v4f*>(pr + 4);
  v4f x2 = *reinterpret_cast<const v4f*>(pr + 16);
  v4f x3 = *reinterpret_cast<const v4f*>(pr + 20);
  v16h f;
#pragma unroll
  for (int i = 0; i < 4; ++i) {
    f[i]      = (half_t)x0[i];
    f[4 + i]  = (half_t)x1[i];
    f[8 + i]  = (half_t)x2[i];
    f[12 + i] = (half_t)x3[i];
  }
  return f;
}

// B fragment (32x16). C = A @ W^T, W stored (N x K) row-major: lane holds
// column n = lane&15, halves 0..7 = K[kb..kb+7], 8..15 = K[16+kb..], kb = 8*(lane>=16).
__device__ __forceinline__ v16h load_frag_b(const half_t* p0, int ld, int lane) {
  const int n  = lane & 15;
  const int kb = (lane >> 4) << 3;            // 0 or 8
  const half_t* p = p0 + (size_t)n * (size_t)ld + kb;
  v8h lo = *reinterpret_cast<const v8h*>(p);
  v8h hi = *reinterpret_cast<const v8h*>(p + 16);
  v16h f;
#pragma unroll
  for (int i = 0; i < 8; ++i) { f[i] = lo[i]; f[i + 8] = hi[i]; }
  return f;
}

__device__ __forceinline__ v16h load_frag_b(const float* p0, int ld, int lane) {
  const int n  = lane & 15;
  const int kb = (lane >> 4) << 3;
  const float* p = p0 + (size_t)n * (size_t)ld + kb;
  v4f x0 = *reinterpret_cast<const v4f*>(p);
  v4f x1 = *reinterpret_cast<const v4f*>(p + 4);
  v4f x2 = *reinterpret_cast<const v4f*>(p + 16);
  v4f x3 = *reinterpret_cast<const v4f*>(p + 20);
  v16h f;
#pragma unroll
  for (int i = 0; i < 4; ++i) {
    f[i]      = (half_t)x0[i];
    f[4 + i]  = (half_t)x1[i];
    f[8 + i]  = (half_t)x2[i];
    f[12 + i] = (half_t)x3[i];
  }
  return f;
}

// ---------------- projection / FC GEMM ------------------------------------
// C(M x N) = A(M x K) @ W(N x K)^T + bias, block tile 128x64, 8 waves,
// each wave a 32x32 tile (2x2 WMMA frags). A may be f32 or f16; W is f32.
enum { EPI_QH = 0, EPI_VT = 1, EPI_FC = 2 };

template <int EPI, typename TA>
__global__ void __launch_bounds__(256)
gemm_proj(const TA* __restrict__ A, const float* __restrict__ W,
          const float* __restrict__ bias, const float* __restrict__ residual,
          half_t* __restrict__ out16, float* __restrict__ out32,
          int M, int N, int K) {
  const int lane = threadIdx.x & 31;
  const int wave = threadIdx.x >> 5;
  const int bm = blockIdx.y * 128 + (wave >> 1) * 32;
  const int bn = blockIdx.x * 64 + (wave & 1) * 32;
  const int mr = lane & 15;

  // Startup prefetch of the first K-tiles (branch-free; loops stay clean).
  __builtin_prefetch(A + (size_t)(bm + mr) * K, 0, 3);
  __builtin_prefetch(A + (size_t)(bm + 16 + mr) * K, 0, 3);
  __builtin_prefetch(W + (size_t)(bn + mr) * K, 0, 3);
  __builtin_prefetch(W + (size_t)(bn + 16 + mr) * K, 0, 3);

  v8f acc[2][2] = {};
#pragma unroll 2
  for (int k = 0; k < K; k += 32) {
    v16h a0 = load_frag_a(A + (size_t)(bm +  0) * K + k, K, lane);
    v16h a1 = load_frag_a(A + (size_t)(bm + 16) * K + k, K, lane);
    v16h b0 = load_frag_b(W + (size_t)(bn +  0) * K + k, K, lane);
    v16h b1 = load_frag_b(W + (size_t)(bn + 16) * K + k, K, lane);
    acc[0][0] = wmma_f16(a0, b0, acc[0][0]);
    acc[0][1] = wmma_f16(a0, b1, acc[0][1]);
    acc[1][0] = wmma_f16(a1, b0, acc[1][0]);
    acc[1][1] = wmma_f16(a1, b1, acc[1][1]);
  }

  const int nlo = lane & 15;
  const int mhi = (lane >> 4) * 8;
#pragma unroll
  for (int tm = 0; tm < 2; ++tm)
#pragma unroll
    for (int tn = 0; tn < 2; ++tn)
#pragma unroll
      for (int r = 0; r < 8; ++r) {
        const int gm = bm + tm * 16 + mhi + r;
        const int gn = bn + tn * 16 + nlo;
        float val = acc[tm][tn][r] + bias[gn];
        if (EPI == EPI_QH) {
          // scatter to [b*H+h][row][dk] head-major layout
          const int b = gm >> 11, l = gm & 2047, h = gn >> 6, d = gn & 63;
          out16[((size_t)(b * H_SZ + h) * L_SZ + l) * DK_SZ + d] = (half_t)val;
        } else if (EPI == EPI_VT) {
          // scatter transposed: [b*H+h][dv][t]
          const int b = gm >> 11, t = gm & 2047, h = gn >> 6, d = gn & 63;
          out16[((size_t)(b * H_SZ + h) * DK_SZ + d) * T_SZ + t] = (half_t)val;
        } else { // EPI_FC: + residual, f32 row-major
          const size_t idx = (size_t)gm * N + gn;
          out32[idx] = val + residual[idx];
        }
      }
}

// ---------------- fused attention ------------------------------------------
// One block per (b,h, 32-row l tile). Score tile (32 x 2048 f32) lives in
// LDS (padded stride). Phase 1: S = Q K^T / 8 via WMMA -> LDS. Phase 2:
// register-resident softmax per row, normalized P -> LDS + attn (single
// HBM write of attn). Phase 3: O = P @ V with P fragments read from LDS.
__global__ void __launch_bounds__(256)
attn_fused(const half_t* __restrict__ qh, const half_t* __restrict__ kh,
           const half_t* __restrict__ vhT, float* __restrict__ attn,
           half_t* __restrict__ o16) {
  __shared__ float smem[TILE_L * SROW];   // ~257 KB of the WGP's 320 KB LDS

  const int bh   = blockIdx.y;
  const int b    = bh >> 4, h = bh & 15;
  const int lblk = blockIdx.x * TILE_L;
  const int lane = threadIdx.x & 31;
  const int wave = threadIdx.x >> 5;
  const int nlo  = lane & 15;
  const int mhi  = (lane >> 4) * 8;

  const half_t* Q  = qh  + ((size_t)bh * L_SZ + lblk) * DK_SZ;
  const half_t* Kh = kh  + (size_t)bh * T_SZ * DK_SZ;
  const half_t* Vt = vhT + (size_t)bh * DK_SZ * T_SZ;

  // Startup prefetch: first K-rows of this wave's t-range and first V rows.
  __builtin_prefetch(Kh + (size_t)(wave * 256 + nlo) * DK_SZ, 0, 3);
  __builtin_prefetch(Vt + (size_t)((wave & 3) * 16 + nlo) * T_SZ, 0, 3);

  // ---- Phase 1: S tile -> LDS. Q fragments loaded once, reused 16x. ----
  v16h a00 = load_frag_a(Q +  0 * DK_SZ +  0, DK_SZ, lane);
  v16h a01 = load_frag_a(Q +  0 * DK_SZ + 32, DK_SZ, lane);
  v16h a10 = load_frag_a(Q + 16 * DK_SZ +  0, DK_SZ, lane);
  v16h a11 = load_frag_a(Q + 16 * DK_SZ + 32, DK_SZ, lane);

  const float scale = 0.125f;  // 1/sqrt(64)
#pragma unroll 2
  for (int tn = 0; tn < 16; ++tn) {
    const int bn = wave * 256 + tn * 16;  // t offset of this 16-col tile
    v16h b0 = load_frag_b(Kh + (size_t)bn * DK_SZ +  0, DK_SZ, lane);
    v16h b1 = load_frag_b(Kh + (size_t)bn * DK_SZ + 32, DK_SZ, lane);
    v8f acc0 = {}, acc1 = {};
    acc0 = wmma_f16(a00, b0, acc0);
    acc0 = wmma_f16(a01, b1, acc0);
    acc1 = wmma_f16(a10, b0, acc1);
    acc1 = wmma_f16(a11, b1, acc1);
#pragma unroll
    for (int r = 0; r < 8; ++r) {
      smem[(mhi + r) * SROW + bn + nlo]      = acc0[r] * scale;
      smem[(16 + mhi + r) * SROW + bn + nlo] = acc1[r] * scale;
    }
  }
  __syncthreads();

  // ---- Phase 2: softmax, each wave owns 4 rows; values in registers ----
  const size_t abase = ((size_t)b * L_SZ * H_SZ + h) * (size_t)T_SZ;
  const size_t arst  = (size_t)H_SZ * T_SZ;  // attn stride between l rows
  for (int rr = 0; rr < 4; ++rr) {
    const int row = wave * 4 + rr;
    float* sp = smem + (size_t)row * SROW;
    float vals[64];
    float m = -1e30f;
#pragma unroll
    for (int j = 0; j < 64; ++j) {
      vals[j] = sp[lane + 32 * j];
      m = fmaxf(m, vals[j]);
    }
#pragma unroll
    for (int off = 16; off > 0; off >>= 1) m = fmaxf(m, __shfl_xor(m, off, 32));
    float s = 0.f;
#pragma unroll
    for (int j = 0; j < 64; ++j) {
      vals[j] = __expf(vals[j] - m);
      s += vals[j];
    }
#pragma unroll
    for (int off = 16; off > 0; off >>= 1) s += __shfl_xor(s, off, 32);
    const float inv = 1.0f / s;
    float* gp = attn + abase + (size_t)(lblk + row) * arst;
#pragma unroll
    for (int j = 0; j < 64; ++j) {
      const float p = vals[j] * inv;
      sp[lane + 32 * j] = p;   // keep for PV
      gp[lane + 32 * j] = p;   // the single mandatory HBM write of attn
    }
  }
  __syncthreads();

  // ---- Phase 3: O = P @ V. Each wave owns one 16x16 output tile. Two
  //      independent accumulators over alternating k-steps for WMMA ILP. ----
  const int tm  = wave >> 2;  // 0..1 : m tile
  const int tn3 = wave & 3;   // 0..3 : n tile (dv)
  const float*  Ap = smem + (size_t)(tm * 16) * SROW;
  const half_t* Wv = Vt + (size_t)(tn3 * 16) * T_SZ;
  const int am  = (lane & 15);
  const int kb8 = (lane >> 4) << 3;

  v8f accA = {}, accB = {};
#pragma unroll 2
  for (int k = 0; k < T_SZ; k += 64) {
    v16h aA = load_frag_a_lds(Ap + (size_t)am * SROW + k + kb8);
    v16h bA = load_frag_b(Wv + k, T_SZ, lane);
    accA = wmma_f16(aA, bA, accA);
    v16h aB = load_frag_a_lds(Ap + (size_t)am * SROW + (k + 32) + kb8);
    v16h bB = load_frag_b(Wv + (k + 32), T_SZ, lane);
    accB = wmma_f16(aB, bB, accB);
  }
#pragma unroll
  for (int r = 0; r < 8; ++r) {
    const int gl = lblk + tm * 16 + mhi + r;
    const int gv = tn3 * 16 + nlo;
    o16[(size_t)(b * L_SZ + gl) * DM_SZ + h * DK_SZ + gv] =
        (half_t)(accA[r] + accB[r]);
  }
}

// ---------------- LayerNorm over D_MODEL -----------------------------------
__global__ void __launch_bounds__(256)
layernorm_rows(const float* __restrict__ x, const float* __restrict__ gamma,
               const float* __restrict__ beta, float* __restrict__ out) {
  const float* p = x + (size_t)blockIdx.x * DM_SZ;
  float* o = out + (size_t)blockIdx.x * DM_SZ;
  const int tid = threadIdx.x;
  __shared__ float red[8], red2[8];

  float s = 0.f, s2 = 0.f;
  for (int i = tid; i < DM_SZ; i += 256) {
    float v = p[i];
    s += v;
    s2 += v * v;
  }
#pragma unroll
  for (int off = 16; off > 0; off >>= 1) {
    s  += __shfl_xor(s, off, 32);
    s2 += __shfl_xor(s2, off, 32);
  }
  if ((tid & 31) == 0) { red[tid >> 5] = s; red2[tid >> 5] = s2; }
  __syncthreads();
  s = 0.f; s2 = 0.f;
#pragma unroll
  for (int w = 0; w < 8; ++w) { s += red[w]; s2 += red2[w]; }
  const float mu   = s * (1.0f / DM_SZ);
  const float var  = s2 * (1.0f / DM_SZ) - mu * mu;
  const float rstd = rsqrtf(var + 1e-5f);
  for (int i = tid; i < DM_SZ; i += 256)
    o[i] = (p[i] - mu) * rstd * gamma[i] + beta[i];
}

// ---------------- host launch ----------------------------------------------

extern "C" void kernel_launch(void* const* d_in, const int* in_sizes, int n_in,
                              void* d_out, int out_size, void* d_ws, size_t ws_size,
                              hipStream_t stream) {
  const float* q    = (const float*)d_in[0];
  const float* k    = (const float*)d_in[1];
  const float* v    = (const float*)d_in[2];
  const float* w_q  = (const float*)d_in[3];
  const float* b_q  = (const float*)d_in[4];
  const float* w_k  = (const float*)d_in[5];
  const float* b_k  = (const float*)d_in[6];
  const float* w_v  = (const float*)d_in[7];
  const float* b_v  = (const float*)d_in[8];
  const float* w_fc = (const float*)d_in[9];
  const float* b_fc = (const float*)d_in[10];
  const float* gamma= (const float*)d_in[11];
  const float* beta = (const float*)d_in[12];

  float* out  = (float*)d_out;                           // [4096][1024]
  float* attn = out + (size_t)MROWS * DM_SZ;             // [B][L][H][T]

  const size_t ACT = (size_t)MROWS * DM_SZ;              // 4194304
  half_t* qh16  = (half_t*)d_ws;        // [32][2048][64]
  half_t* kh16  = qh16  + ACT;          // [32][2048][64]
  half_t* vhT16 = kh16  + ACT;          // [32][64][2048]
  half_t* o16   = vhT16 + ACT;          // [4096][1024]
  float*  xbuf  = (float*)(o16 + ACT);  // [4096][1024] f32

  const dim3 blk(256);

  // 1) QKV projections straight from f32 inputs (M=4096, N=1024, K=1024)
  gemm_proj<EPI_QH, float><<<dim3(16, 32, 1), blk, 0, stream>>>(
      q, w_q, b_q, nullptr, qh16, nullptr, MROWS, DM_SZ, DM_SZ);
  gemm_proj<EPI_QH, float><<<dim3(16, 32, 1), blk, 0, stream>>>(
      k, w_k, b_k, nullptr, kh16, nullptr, MROWS, DM_SZ, DM_SZ);
  gemm_proj<EPI_VT, float><<<dim3(16, 32, 1), blk, 0, stream>>>(
      v, w_v, b_v, nullptr, vhT16, nullptr, MROWS, DM_SZ, DM_SZ);

  // 2) fused QK^T -> softmax -> PV; attn written exactly once
  attn_fused<<<dim3(L_SZ / TILE_L, B_SZ * H_SZ), blk, 0, stream>>>(
      qh16, kh16, vhT16, attn, o16);

  // 3) FC + bias + residual
  gemm_proj<EPI_FC, half_t><<<dim3(16, 32, 1), blk, 0, stream>>>(
      o16, w_fc, b_fc, q, nullptr, xbuf, MROWS, DM_SZ, DM_SZ);

  // 4) LayerNorm -> d_out
  layernorm_rows<<<dim3(MROWS), blk, 0, stream>>>(xbuf, gamma, beta, out);
}
